// CVA_Attention_55198919688796
// MI455X (gfx1250) — matrix operand order repaired and checked by reading, not compiled, not run on hardware
//
#include <hip/hip_runtime.h>
#include <hip/hip_bf16.h>

typedef __attribute__((ext_vector_type(16))) _Float16 v16h;
typedef __attribute__((ext_vector_type(8)))  float    v8f;
typedef unsigned int u32;
typedef __attribute__((ext_vector_type(4))) u32 u32x4;
typedef __attribute__((ext_vector_type(8))) u32 u32x8;
typedef _Float16 h16;

#define CC      256   // channels
#define BB      8     // batch
#define NN      4096  // H*W
#define BN      32768 // BB*NN
#define NH      8
#define HD      32
#define CHK     64
#define NCK     64    // NN/CHK
#define LNEPS   1e-5f
#define A_LD    40    // padded LDS stride (halves) for 32-wide tiles
#define P_LD    72    // padded LDS stride (halves) for 64-wide tiles
#define AF_LD   264   // padded LDS stride (halves) for 256-wide panels

// ---------------------------------------------------------------------------
// CDNA5 async copy: per-lane 16B global -> LDS (ASYNCcnt tracked)
__device__ __forceinline__ void async_b128(const void* g, void* l) {
    u32 lo = (u32)(uintptr_t)l;   // LDS generic addr truncates to LDS offset
    asm volatile("global_load_async_to_lds_b128 %0, %1, off"
                 :: "v"(lo), "v"(g) : "memory");
}
__device__ __forceinline__ void async_wait0() {
    asm volatile("s_wait_asynccnt 0x0" ::: "memory");
}

// CDNA5 Tensor Data Mover: ROWS x DIM0 f16 tile (contiguous rows of DIM0
// elements) -> LDS, HW-padded by 16B after every row.
// PAD_IVL: pad threshold code, row bytes = 8 << PAD_IVL.
// 2-group descriptor form (<=2D tensor). Wave-level op, EXEC ignored.
template <int DIM0, int ROWS, int PAD_IVL>
__device__ __forceinline__ void tdm_load_tile(const h16* gsrc, void* ldsdst) {
    unsigned long long ga = (unsigned long long)(uintptr_t)gsrc;
    u32x4 g0;
    g0[0] = 1u;                                            // count=1, user D#
    g0[1] = (u32)(uintptr_t)ldsdst;                        // lds_addr (bytes)
    g0[2] = (u32)ga;                                       // global_addr[31:0]
    g0[3] = ((u32)(ga >> 32) & 0x01FFFFFFu) | (2u << 30);  // addr[56:32] | type=2
    u32x8 g1 = {};
    g1[0] = (1u << 16)                 // data_size = 2 bytes
          | (1u << 20)                 // pad_enable
          | ((u32)PAD_IVL << 22)       // pad after each row
          | (3u << 25);                // pad_amount: 4 DWORDs (16B)
    g1[1] = (u32)DIM0 << 16;           // tensor_dim0 (bits 79:48)
    g1[2] = (u32)ROWS << 16;           // tensor_dim1 (bits 111:80)
    g1[3] = (u32)DIM0 << 16;           // tile_dim0   (bits 127:112)
    g1[4] = (u32)ROWS;                 // tile_dim1   (bits 143:128)
    g1[5] = (u32)DIM0;                 // tensor_dim0_stride (bits 207:160)
    asm volatile("tensor_load_to_lds %0, %1" :: "s"(g0), "s"(g1) : "memory");
}

// ---------------------------------------------------------------------------
// Fragment builders (CDNA5 ISA 7.12.2, wave32)
__device__ __forceinline__ v16h load_a_frag(const h16* base, int stride, int lane) {
    int m  = lane & 15;
    int kb = (lane >> 4) * 8;
    const h16* p0 = base + m * stride + kb;
    const h16* p1 = p0 + 16;
    v16h a;
#pragma unroll
    for (int i = 0; i < 8; ++i) { a[i] = p0[i]; a[8 + i] = p1[i]; }
    return a;
}

__device__ __forceinline__ v16h load_b_frag(const h16* base, int stride, int lane) {
    int n  = lane & 15;
    int kb = (lane >> 4) * 8;
    const h16* p = base + n * stride + kb;
    v16h b;
#pragma unroll
    for (int i = 0; i < 8; ++i) { b[i] = p[i]; b[8 + i] = p[16 + i]; }
    return b;
}

__device__ __forceinline__ v8f wmma_f16(v16h a, v16h b, v8f c) {
    return __builtin_amdgcn_wmma_f32_16x16x32_f16(false, a, false, b, (short)0, c,
                                                  false, false);
}

// ---------------------------------------------------------------------------
// Kernel 1: convert 4 weight matrices f32 [cin][cout] -> f16 transposed [w][cout][cin]
__global__ __launch_bounds__(256) void wconv(const float* __restrict__ Wq,
                                             const float* __restrict__ Wk,
                                             const float* __restrict__ Wv,
                                             const float* __restrict__ Wp,
                                             h16* __restrict__ wT) {
    int idx  = blockIdx.x * 256 + threadIdx.x;
    int w    = idx >> 16;
    int e    = idx & 65535;
    int cout = e >> 8;
    int cin  = e & 255;
    const float* W = (w == 0) ? Wq : (w == 1) ? Wk : (w == 2) ? Wv : Wp;
    wT[(size_t)w * 65536 + e] = (h16)W[cin * CC + cout];
}

// ---------------------------------------------------------------------------
// Kernel 2a: LN stats for contiguous rows (CVA): one wave per row
__global__ __launch_bounds__(256) void ln_stats_rows(const float* __restrict__ src,
                                                     float* __restrict__ mean,
                                                     float* __restrict__ rstd) {
    int wave = threadIdx.x >> 5, lane = threadIdx.x & 31;
    size_t row = (size_t)blockIdx.x * 8 + wave;
    const float* p = src + row * CC;
    float s = 0.f, s2 = 0.f;
#pragma unroll
    for (int j = 0; j < 8; ++j) {
        float v = p[j * 32 + lane];
        s += v; s2 += v * v;
    }
#pragma unroll
    for (int m = 16; m >= 1; m >>= 1) {
        s  += __shfl_xor(s, m, 32);
        s2 += __shfl_xor(s2, m, 32);
    }
    if (lane == 0) {
        float mu  = s * (1.f / CC);
        float var = s2 * (1.f / CC) - mu * mu;
        mean[row] = mu;
        rstd[row] = rsqrtf(var + LNEPS);
    }
}

// Kernel 2b: LN stats for xT rows: xs[b,n,c] = x[b,c,n]; block handles 64 rows
__global__ __launch_bounds__(256) void ln_stats_x(const float* __restrict__ x,
                                                  float* __restrict__ mean,
                                                  float* __restrict__ rstd) {
    size_t row0 = (size_t)blockIdx.x * 64;
    int b  = (int)(row0 >> 12);
    int n0 = (int)(row0 & 4095);
    int tx = threadIdx.x & 63;
    int ty = threadIdx.x >> 6;
    float s = 0.f, s2 = 0.f;
    const float* xb = x + (size_t)b * CC * NN;
    for (int c = ty; c < CC; c += 4) {
        float v = xb[(size_t)c * NN + n0 + tx];
        s += v; s2 += v * v;
    }
    __shared__ float ps[4][64], ps2[4][64];
    ps[ty][tx] = s; ps2[ty][tx] = s2;
    __syncthreads();
    if (ty == 0) {
        float S  = ps[0][tx] + ps[1][tx] + ps[2][tx] + ps[3][tx];
        float S2 = ps2[0][tx] + ps2[1][tx] + ps2[2][tx] + ps2[3][tx];
        float mu = S * (1.f / CC);
        mean[row0 + tx] = mu;
        rstd[row0 + tx] = rsqrtf(S2 * (1.f / CC) - mu * mu + LNEPS);
    }
}

// ---------------------------------------------------------------------------
// Kernel 3: fused LN + QKV projection, one 64-row block x all 8 heads per WG.
// grid = (BN/64, 3), block = 128 (4 waves). A panel staged once (LN fused),
// per-head weight panel async-staged; inner K loop is barrier-free.
// Q/K stored [b][h][chunk][pos][d]; V stored transposed [b][h][chunk][d][pos].
__global__ __launch_bounds__(128) void qkv_gemm(
    const float* __restrict__ cva, const float* __restrict__ x,
    const float* __restrict__ mean_c, const float* __restrict__ rstd_c,
    const float* __restrict__ mean_x, const float* __restrict__ rstd_x,
    const float* __restrict__ qg, const float* __restrict__ qb,
    const float* __restrict__ kg, const float* __restrict__ kb,
    const float* __restrict__ vg, const float* __restrict__ vb,
    const h16* __restrict__ wT,
    const float* __restrict__ bq, const float* __restrict__ bk,
    const float* __restrict__ bv,
    h16* __restrict__ Qb, h16* __restrict__ Kb, h16* __restrict__ Vb) {
    const int z = blockIdx.y;             // 0=Q 1=K 2=V
    const size_t row0 = (size_t)blockIdx.x * 64;
    const int bi = (int)(row0 >> 12);
    const int n0 = (int)(row0 & 4095);
    const int chunk = n0 >> 6;

    const float* mean = (z == 2) ? mean_x : mean_c;
    const float* rstd = (z == 2) ? rstd_x : rstd_c;
    const float* g    = (z == 0) ? qg : (z == 1) ? kg : vg;
    const float* be   = (z == 0) ? qb : (z == 1) ? kb : vb;
    const float* bias = (z == 0) ? bq : (z == 1) ? bk : bv;
    const h16*   wt   = wT + (size_t)z * 65536;
    h16*         outp = (z == 0) ? Qb : (z == 1) ? Kb : Vb;

    __shared__ h16 sA[64 * AF_LD];     // LN'd activation panel 64x256
    __shared__ h16 sBh[32 * AF_LD];    // one head's weight panel 32x256
    __shared__ float sMu[64], sRs[64];
    __shared__ float sG[CC], sE[CC];

    const int t = threadIdx.x, wave = t >> 5, lane = t & 31;

    if (t < 64) { sMu[t] = mean[row0 + t]; sRs[t] = rstd[row0 + t]; }
    sG[t] = g[t]; sG[t + 128] = g[t + 128];
    sE[t] = be[t]; sE[t + 128] = be[t + 128];
    __syncthreads();

    // ---- stage full A panel with LN fused (once per block) ----
    if (z < 2) {
        int row = t >> 1;
        int cb  = (t & 1) * 128;
        float mu = sMu[row], rs = sRs[row];
        const float* p = cva + (row0 + row) * CC + cb;
#pragma unroll 16
        for (int j = 0; j < 128; ++j) {
            int kix = cb + j;
            sA[row * AF_LD + kix] = (h16)((p[j] - mu) * rs * sG[kix] + sE[kix]);
        }
    } else {
        for (int cc = t; cc < CC; cc += 128) {
            const float* xp = x + (size_t)bi * CC * NN + (size_t)cc * NN + n0;
            float gk = sG[cc], ek = sE[cc];
#pragma unroll 8
            for (int row = 0; row < 64; ++row)
                sA[row * AF_LD + cc] = (h16)((xp[row] - sMu[row]) * sRs[row] * gk + ek);
        }
    }

    const int ncol = lane & 15;
    for (int h = 0; h < NH; ++h) {
        // ---- async-stage this head's weight rows: 32 x 256 halves ----
        {
            int r  = t >> 2;             // cout offset 0..31
            int cb = (t & 3) * 64;       // 64 halves = 128B
            const h16* wr = wt + (size_t)(h * HD + r) * CC + cb;
#pragma unroll
            for (int i = 0; i < 8; ++i)
                async_b128(wr + 8 * i, &sBh[r * AF_LD + cb + 8 * i]);
        }
        async_wait0();
        __syncthreads();                 // sBh ready (and sA on first pass)

        v8f acc0 = {}, acc1 = {};
#pragma unroll
        for (int kk = 0; kk < CC; kk += 32) {
            v16h af  = load_a_frag(sA + wave * 16 * AF_LD + kk, AF_LD, lane);
            v16h bf0 = load_b_frag(sBh + kk, AF_LD, lane);
            v16h bf1 = load_b_frag(sBh + 16 * AF_LD + kk, AF_LD, lane);
            acc0 = wmma_f16(af, bf0, acc0);
            acc1 = wmma_f16(af, bf1, acc1);
        }

        // ---- epilogue: bias, convert, store ----
        const size_t tbase = ((((size_t)bi * NH + h) * NCK + chunk) * 64) * HD;
        float b0 = bias[h * HD + ncol], b1 = bias[h * HD + 16 + ncol];
#pragma unroll
        for (int r = 0; r < 8; ++r) {
            int pos = wave * 16 + r + ((lane >> 4) << 3);
            if (z == 2) {   // V stored transposed: [d][pos]
                outp[tbase + (size_t)ncol * 64 + pos]        = (h16)(acc0[r] + b0);
                outp[tbase + (size_t)(ncol + 16) * 64 + pos] = (h16)(acc1[r] + b1);
            } else {        // Q/K row-major: [pos][d]
                outp[tbase + (size_t)pos * HD + ncol]      = (h16)(acc0[r] + b0);
                outp[tbase + (size_t)pos * HD + 16 + ncol] = (h16)(acc1[r] + b1);
            }
        }
        __syncthreads();                 // before next head overwrites sBh
    }
}

// ---------------------------------------------------------------------------
// Kernel 4: block-local attention per (chunk, head, batch). block = 128 (4 waves).
// Q, K, V^T tiles all arrive via the Tensor Data Mover with HW row padding.
__global__ __launch_bounds__(128) void attn(const h16* __restrict__ Qb,
                                            const h16* __restrict__ Kb,
                                            const h16* __restrict__ Vb,
                                            h16* __restrict__ Ob) {
    const int chunk = blockIdx.x, head = blockIdx.y, bi = blockIdx.z;
    const size_t tbase = ((((size_t)bi * NH + head) * NCK + chunk) * 64) * HD;

    __shared__ h16 sq[64 * A_LD];
    __shared__ h16 sk[64 * A_LD];
    __shared__ h16 svt[32 * P_LD];   // V transposed: [d][key]
    __shared__ h16 sp[64 * P_LD];    // probabilities f16

    const int t = threadIdx.x, wave = t >> 5, lane = t & 31;

    if (wave == 0) {                 // one wave issues all three tile DMAs
        tdm_load_tile<32, 64, 3>(Qb + tbase, sq);   // 64B rows -> 80B stride
        tdm_load_tile<32, 64, 3>(Kb + tbase, sk);
        tdm_load_tile<64, 32, 4>(Vb + tbase, svt);  // 128B rows -> 144B stride
    }
    __builtin_amdgcn_s_wait_tensorcnt(0);
    __syncthreads();

    // scores: wave w handles query rows w*16..w*16+15, all 64 keys
    v16h aq = load_a_frag(sq + wave * 16 * A_LD, A_LD, lane);
    v8f s[4];
#pragma unroll
    for (int nt = 0; nt < 4; ++nt) {
        v8f zero = {};
        v16h bk = load_b_frag(sk + nt * 16 * A_LD, A_LD, lane);
        s[nt] = wmma_f16(aq, bk, zero);
    }

    // softmax over each row (64 cols = 4 accums x 16 lanes of this half)
    const float scale = 0.17677669529663687f;  // 1/sqrt(32)
#pragma unroll
    for (int r = 0; r < 8; ++r) {
        float mx = -1e30f;
#pragma unroll
        for (int nt = 0; nt < 4; ++nt) { s[nt][r] *= scale; mx = fmaxf(mx, s[nt][r]); }
#pragma unroll
        for (int m = 8; m >= 1; m >>= 1) mx = fmaxf(mx, __shfl_xor(mx, m, 32));
        float sum = 0.f;
#pragma unroll
        for (int nt = 0; nt < 4; ++nt) {
            float e = __expf(s[nt][r] - mx);
            s[nt][r] = e; sum += e;
        }
#pragma unroll
        for (int m = 8; m >= 1; m >>= 1) sum += __shfl_xor(sum, m, 32);
        float rinv = 1.f / sum;
        int row = wave * 16 + r + ((lane >> 4) << 3);
#pragma unroll
        for (int nt = 0; nt < 4; ++nt)
            sp[row * P_LD + nt * 16 + (lane & 15)] = (h16)(s[nt][r] * rinv);
    }
    __syncthreads();

    // O = P x V : M=16 per wave, N=32 (2 tiles), K=64 (2 x WMMA-K32)
    v8f o0 = {}, o1 = {};
#pragma unroll
    for (int kc = 0; kc < 64; kc += 32) {
        v16h ap  = load_a_frag(sp + wave * 16 * P_LD + kc, P_LD, lane);
        v16h bv0 = load_b_frag(svt + kc, P_LD, lane);
        v16h bv1 = load_b_frag(svt + 16 * P_LD + kc, P_LD, lane);
        o0 = wmma_f16(ap, bv0, o0);
        o1 = wmma_f16(ap, bv1, o1);
    }

    const int n0 = chunk * 64;
#pragma unroll
    for (int r = 0; r < 8; ++r) {
        int pos = wave * 16 + r + ((lane >> 4) << 3);
        size_t grow = (size_t)bi * NN + n0 + pos;
        Ob[grow * CC + head * HD + (lane & 15)]      = (h16)o0[r];
        Ob[grow * CC + head * HD + 16 + (lane & 15)] = (h16)o1[r];
    }
}

// ---------------------------------------------------------------------------
// Kernel 5: output projection + transpose store to [B,C,H,W] f32.
// grid = BN/64, block = 128 (4 waves). O panel async-staged once; 8 col tiles.
__global__ __launch_bounds__(128) void proj(const h16* __restrict__ Ob,
                                            const h16* __restrict__ wpT,
                                            const float* __restrict__ bp,
                                            float* __restrict__ out) {
    const size_t row0 = (size_t)blockIdx.x * 64;
    const int bi = (int)(row0 >> 12);
    const int n0 = (int)(row0 & 4095);

    __shared__ h16 sA[64 * AF_LD];
    __shared__ h16 sBp[32 * AF_LD];
    __shared__ float sO[64 * 33];

    const int t = threadIdx.x, wave = t >> 5, lane = t & 31;

    {   // async-stage full O panel: 64 x 256 halves
        int row = t >> 1;
        int cb  = (t & 1) * 128;
        const h16* p = Ob + (row0 + row) * CC + cb;
#pragma unroll
        for (int i = 0; i < 16; ++i)
            async_b128(p + 8 * i, &sA[row * AF_LD + cb + 8 * i]);
    }

    for (int ct = 0; ct < 8; ++ct) {
        {   // async-stage this col tile's weight rows: 32 x 256 halves
            int r  = t >> 2;
            int cb = (t & 3) * 64;
            const h16* wr = wpT + (size_t)(ct * 32 + r) * CC + cb;
#pragma unroll
            for (int i = 0; i < 8; ++i)
                async_b128(wr + 8 * i, &sBp[r * AF_LD + cb + 8 * i]);
        }
        async_wait0();                  // covers sA panel too on first pass
        __syncthreads();

        v8f acc0 = {}, acc1 = {};
#pragma unroll
        for (int kk = 0; kk < CC; kk += 32) {
            v16h af  = load_a_frag(sA + wave * 16 * AF_LD + kk, AF_LD, lane);
            v16h bf0 = load_b_frag(sBp + kk, AF_LD, lane);
            v16h bf1 = load_b_frag(sBp + 16 * AF_LD + kk, AF_LD, lane);
            acc0 = wmma_f16(af, bf0, acc0);
            acc1 = wmma_f16(af, bf1, acc1);
        }

        float b0 = bp[ct * 32 + (lane & 15)];
        float b1 = bp[ct * 32 + 16 + (lane & 15)];
#pragma unroll
        for (int r = 0; r < 8; ++r) {
            int row = wave * 16 + r + ((lane >> 4) << 3);
            sO[row * 33 + (lane & 15)]      = acc0[r] + b0;
            sO[row * 33 + 16 + (lane & 15)] = acc1[r] + b1;
        }
        __syncthreads();

        {   // out[b, c, n] = tile[n - n0, c - ct*32]; contiguous along n
            int cc = t >> 2;
            int rb = (t & 3) * 16;
            float* po = out + (size_t)bi * CC * NN + (size_t)(ct * 32 + cc) * NN
                      + n0 + rb;
#pragma unroll
            for (int j = 0; j < 16; ++j) po[j] = sO[(rb + j) * 33 + cc];
        }
        __syncthreads();               // before next ct overwrites sBp / sO
    }
}

// ---------------------------------------------------------------------------
extern "C" void kernel_launch(void* const* d_in, const int* in_sizes, int n_in,
                              void* d_out, int out_size, void* d_ws, size_t ws_size,
                              hipStream_t stream) {
    (void)in_sizes; (void)n_in; (void)out_size; (void)ws_size;
    const float* x    = (const float*)d_in[0];
    const float* cva  = (const float*)d_in[1];
    const float* qn_g = (const float*)d_in[2];
    const float* qn_b = (const float*)d_in[3];
    const float* kn_g = (const float*)d_in[4];
    const float* kn_b = (const float*)d_in[5];
    const float* vn_g = (const float*)d_in[6];
    const float* vn_b = (const float*)d_in[7];
    const float* Wq   = (const float*)d_in[8];
    const float* bq   = (const float*)d_in[9];
    const float* Wk   = (const float*)d_in[10];
    const float* bk   = (const float*)d_in[11];
    const float* Wv   = (const float*)d_in[12];
    const float* bv   = (const float*)d_in[13];
    const float* Wp   = (const float*)d_in[14];
    const float* bp   = (const float*)d_in[15];

    char* ws = (char*)d_ws;
    size_t off = 0;
    auto take = [&](size_t bytes) {
        size_t o = off;
        off = (off + bytes + 255) & ~(size_t)255;
        return o;
    };
    h16*   wT     = (h16*)(ws + take(4 * 65536 * sizeof(h16)));
    float* mean_c = (float*)(ws + take((size_t)BN * 4));
    float* rstd_c = (float*)(ws + take((size_t)BN * 4));
    float* mean_x = (float*)(ws + take((size_t)BN * 4));
    float* rstd_x = (float*)(ws + take((size_t)BN * 4));
    h16*   Qbuf   = (h16*)(ws + take((size_t)BN * CC * sizeof(h16)));
    h16*   Kbuf   = (h16*)(ws + take((size_t)BN * CC * sizeof(h16)));
    h16*   Vbuf   = (h16*)(ws + take((size_t)BN * CC * sizeof(h16)));
    h16*   Obuf   = (h16*)(ws + take((size_t)BN * CC * sizeof(h16)));

    wconv<<<dim3(4 * 65536 / 256), dim3(256), 0, stream>>>(Wq, Wk, Wv, Wp, wT);
    ln_stats_rows<<<dim3(BN / 8), dim3(256), 0, stream>>>(cva, mean_c, rstd_c);
    ln_stats_x<<<dim3(BN / 64), dim3(256), 0, stream>>>(x, mean_x, rstd_x);
    qkv_gemm<<<dim3(BN / 64, 3), dim3(128), 0, stream>>>(
        cva, x, mean_c, rstd_c, mean_x, rstd_x,
        qn_g, qn_b, kn_g, kn_b, vn_g, vn_b,
        wT, bq, bk, bv, Qbuf, Kbuf, Vbuf);
    attn<<<dim3(NCK, NH, BB), dim3(128), 0, stream>>>(Qbuf, Kbuf, Vbuf, Obuf);
    proj<<<dim3(BN / 64), dim3(128), 0, stream>>>(
        Obuf, wT + (size_t)3 * 65536, bp, (float*)d_out);
}
